// GraphEncoderDecoderAttention_11484742549821
// MI455X (gfx1250) — matrix operand order repaired and checked by reading, not compiled, not run on hardware
//
#include <hip/hip_runtime.h>
#include <cmath>

#define DEV __device__ __forceinline__

constexpr int B_ = 2, N_ = 2048, C_ = 256, H_ = 8, FH_ = 32, DH_ = 32;
constexpr float ALPHA_ = 0.1f;
constexpr float NSLOPE_ = 0.01f;
constexpr float NEG_BIG_ = -9.0e15f;
constexpr float EPS_ = 1e-6f;

typedef __attribute__((ext_vector_type(16))) __bf16 v16bf;
typedef __attribute__((ext_vector_type(8)))  float  v8f;

// TDM availability (device pass on both probed toolchains; host pass -> 0)
#if defined(__HIP_DEVICE_COMPILE__) && defined(__has_builtin)
#if __has_builtin(__builtin_amdgcn_tensor_load_to_lds) && __has_builtin(__builtin_amdgcn_s_wait_tensorcnt)
#define HAVE_TDM 1
#else
#define HAVE_TDM 0
#endif
#else
#define HAVE_TDM 0
#endif

// ---------------------------------------------------------------- helpers ---

DEV __bf16 f2bf(float f) { return (__bf16)f; }   // hardware cvt, RNE

DEV v8f wmma_bf16(v16bf a, v16bf b, v8f c) {
  // v_wmma_f32_16x16x32_bf16  D = A(16x32) * B(32x16) + C, fp32 accumulate
  return __builtin_amdgcn_wmma_f32_16x16x32_bf16(false, a, false, b, (short)0, c, false, false);
}

// A operand, source row-major [m][k] fp32.  Lane L: m = L&15, hi = L>>4.
// ISA A-layout: K(e,hi) = e<8 ? e+8*hi : 16+(e-8)+8*hi  -> two contiguous 8-runs.
DEV v16bf ldA_f32(const float* base, int row0, int ld, int k0, int lane) {
  int m = lane & 15, hi = (lane >> 4) & 1;
  const float* p = base + (long)(row0 + m) * ld + k0 + 8 * hi;
  v16bf r;
#pragma unroll
  for (int e = 0; e < 8; ++e) r[e] = f2bf(p[e]);
#pragma unroll
  for (int e = 0; e < 8; ++e) r[8 + e] = f2bf(p[16 + e]);
  return r;
}

DEV v16bf ldA_bf16(const __bf16* base, int row0, int ld, int k0, int lane) {
  int m = lane & 15, hi = (lane >> 4) & 1;
  const __bf16* p = base + (long)(row0 + m) * ld + k0 + 8 * hi;
  v16bf r;
#pragma unroll
  for (int e = 0; e < 8; ++e) r[e] = p[e];
#pragma unroll
  for (int e = 0; e < 8; ++e) r[8 + e] = p[16 + e];
  return r;
}

// B operand: lane holds one COLUMN of the 32x16 B.  Source row-major [n][k].
// ISA B-layout: K(e,hi) = e<8 ? e+8*hi : 16+(e-8)+8*hi  -> two contiguous 8-runs.
DEV v16bf ldB_f32(const float* base, int row0, int ld, int k0, int lane) {
  int m = lane & 15, hi = (lane >> 4) & 1;
  const float* p = base + (long)(row0 + m) * ld + k0 + 8 * hi;
  v16bf r;
#pragma unroll
  for (int e = 0; e < 16; ++e) r[e] = f2bf(p[e < 8 ? e : e + 8]);
  return r;
}

DEV v16bf ldB_bf16(const __bf16* base, int row0, int ld, int k0, int lane) {
  int m = lane & 15, hi = (lane >> 4) & 1;
  const __bf16* p = base + (long)(row0 + m) * ld + k0 + 8 * hi;
  v16bf r;
#pragma unroll
  for (int e = 0; e < 16; ++e) r[e] = p[e < 8 ? e : e + 8];
  return r;
}

// B operand from a k-major [k][n] fp32 matrix (column gather), used for Ws.
DEV v16bf ldB_kmajor_f32(const float* base, int k0, int n0, int ld, int lane) {
  int n = lane & 15, hi = (lane >> 4) & 1;
  v16bf r;
#pragma unroll
  for (int e = 0; e < 16; ++e)
    r[e] = f2bf(base[(long)(k0 + (e < 8 ? e : e + 8) + 8 * hi) * ld + n0 + n]);
  return r;
}

DEV float blockReduceSum256(float v, float* red) {
  int t = threadIdx.x;
  red[t] = v; __syncthreads();
#pragma unroll
  for (int s = 128; s > 0; s >>= 1) { if (t < s) red[t] += red[t + s]; __syncthreads(); }
  float r = red[0]; __syncthreads();
  return r;
}

DEV float blockReduceMax256(float v, float* red) {
  int t = threadIdx.x;
  red[t] = v; __syncthreads();
#pragma unroll
  for (int s = 128; s > 0; s >>= 1) { if (t < s) red[t] = fmaxf(red[t], red[t + s]); __syncthreads(); }
  float r = red[0]; __syncthreads();
  return r;
}

DEV float halfMax16(float x) {
#pragma unroll
  for (int m = 1; m < 16; m <<= 1) x = fmaxf(x, __shfl_xor(x, m, 16));
  return x;
}
DEV float halfSum16(float x) {
#pragma unroll
  for (int m = 1; m < 16; m <<= 1) x += __shfl_xor(x, m, 16);
  return x;
}

// ------------------------------------------------------------ TDM helpers ---

#if HAVE_TDM
typedef __attribute__((ext_vector_type(4))) unsigned tdm_u32x4;
typedef __attribute__((ext_vector_type(8))) int      tdm_i32x8;
typedef __attribute__((ext_vector_type(4))) int      tdm_i32x4;

DEV unsigned lds_addr_of(const void* p) {
  // generic -> LDS addrspace, then take the 32-bit LDS byte offset
  return (unsigned)(unsigned long long)(__attribute__((address_space(3))) const void*)p;
}

// 2D bf16 tile load: tile (tileRows x tileCols) from a row-major tensor with
// row stride rowStride (elements), into contiguous LDS rows of tileCols.
// D# per CDNA5 ISA ch.8: group0 = count|lds_addr|global_addr|type=2,
// group1 packs data_size=2B, tensor_dim0/1, tile_dim0/1, tensor_dim0_stride.
DEV void tdm_load_2d_bf16(unsigned ldsAddr, const __bf16* gptr,
                          unsigned tileCols, unsigned tileRows,
                          unsigned tensorCols, unsigned tensorRows,
                          unsigned rowStride) {
  unsigned long long ga = (unsigned long long)gptr;
  tdm_u32x4 g0 = {
      1u,                                                  // count=1, user D#
      ldsAddr,                                             // lds_addr (bytes)
      (unsigned)(ga & 0xFFFFFFFFu),                        // global_addr[31:0]
      (unsigned)((ga >> 32) & 0x01FFFFFFu) | (2u << 30)    // addr[56:32], type=2
  };
  tdm_i32x8 g1 = {
      (int)(1u << 16),                                     // data_size=1 (2 bytes)
      (int)((tensorCols & 0xFFFFu) << 16),                 // tensor_dim0 lo16 @bit48
      (int)((tensorCols >> 16) | ((tensorRows & 0xFFFFu) << 16)),  // dim0 hi / dim1 lo
      (int)((tensorRows >> 16) | (tileCols << 16)),        // dim1 hi / tile_dim0
      (int)(tileRows & 0xFFFFu),                           // tile_dim1 (tile_dim2=0)
      (int)rowStride,                                      // tensor_dim0_stride lo32
      0, 0
  };
  tdm_i32x4 z4 = {};
#if __clang_major__ >= 23
  tdm_i32x8 z8 = {};
  __builtin_amdgcn_tensor_load_to_lds(g0, g1, z4, z4, z8, 0);
#else
  __builtin_amdgcn_tensor_load_to_lds(g0, g1, z4, z4, 0);
#endif
}
#else
// fallback: cooperative 16B vector copies, 128 threads, one 32x32 bf16 tile
DEV void coop_copy_tile32(__bf16* dst, const __bf16* src, int srcLd) {
  int t = threadIdx.x;
  int r = t >> 2, s = (t & 3) * 8;
  *reinterpret_cast<uint4*>(dst + r * 32 + s) =
      *reinterpret_cast<const uint4*>(src + (long)r * srcLd + s);
}
#endif

// ---------------------------------------------------------------- kernels ---

// wsa[h][c] = sum_f Ws[h][c][f]*a1[h][f] ; wta from Wt,a2.  Rank-1 fold of `a`.
__global__ void fold_a_kernel(const float* Ws, const float* Wt, const float* a,
                              float* wsa, float* wta) {
  int t = blockIdx.x * blockDim.x + threadIdx.x;
  if (t >= H_ * C_) return;
  int h = t >> 8, c = t & (C_ - 1);
  const float* wsrow = Ws + ((long)h * C_ + c) * FH_;
  const float* wtrow = Wt + ((long)h * C_ + c) * FH_;
  const float* a1 = a + (long)h * 2 * FH_;
  const float* a2 = a1 + FH_;
  float s1 = 0.f, s2 = 0.f;
#pragma unroll
  for (int f = 0; f < FH_; ++f) { s1 += wsrow[f] * a1[f]; s2 += wtrow[f] * a2[f]; }
  wsa[t] = s1; wta[t] = s2;
}

// s_ctx[b,h,n] = ctx_with_pos[b,n,:]·wsa[h] ; s_h[b,h,n] = x[b,n,:]·wta[h]
__global__ void score_vec_kernel(const float* ctxp, const float* x,
                                 const float* wsa, const float* wta,
                                 float* sctx, float* sh) {
  int idx = blockIdx.x * blockDim.x + threadIdx.x;
  if (idx >= B_ * H_ * N_) return;
  int b = idx / (H_ * N_);
  int r = idx - b * H_ * N_;
  int h = r / N_, n = r - h * N_;
  const float* rowC = ctxp + ((long)b * N_ + n) * C_;
  const float* rowX = x    + ((long)b * N_ + n) * C_;
  const float* va = wsa + (long)h * C_;
  const float* vb = wta + (long)h * C_;
  float s1 = 0.f, s2 = 0.f;
  for (int c = 0; c < C_; ++c) { s1 += rowC[c] * va[c]; s2 += rowX[c] * vb[c]; }
  sctx[idx] = s1; sh[idx] = s2;
}

// Ws_ctx = ctx_with_pos @ Ws[h], written TRANSPOSED per head as bf16:
// wsctxT[bh][f][n].  One wave per 16x16 tile, 8 wmma over K=256.
__global__ void gat_proj_kernel(const float* ctxp, const float* Ws, __bf16* wsctxT) {
  int bh = blockIdx.y; int b = bh / H_, h = bh % H_;
  int tile = blockIdx.x;
  int n0 = (tile >> 1) * 16, f0 = (tile & 1) * 16;
  int lane = threadIdx.x & 31;
  const float* A  = ctxp + (long)b * N_ * C_;
  const float* Bm = Ws   + (long)h * C_ * FH_;
  v8f c = {};
  for (int k0 = 0; k0 < C_; k0 += 32) {
    v16bf a  = ldA_f32(A, n0, C_, k0, lane);
    v16bf bt = ldB_kmajor_f32(Bm, k0, f0, FH_, lane);
    c = wmma_bf16(a, bt, c);
  }
  // D(M=node,N=f) -> transposed store: row f, contiguous 8 cols of n.
  int f = f0 + (lane & 15);
  int nb = n0 + ((lane >> 4) & 1) * 8;
  __bf16* dst = wsctxT + ((long)bh * FH_ + f) * N_ + nb;
#pragma unroll
  for (int v = 0; v < 8; ++v) dst[v] = f2bf(c[v]);
}

// Per (b,i): row max + denom of the rank-1 masked GAT logits, all 8 heads.
__global__ void gat_ml_kernel(const int* adj, const float* sctx, const float* sh,
                              float* mg, float* lg) {
  int b = blockIdx.y, i = blockIdx.x, t = threadIdx.x;
  __shared__ float red[256];
  __shared__ float mSh[H_];
  const int* adjRow = adj + ((long)b * N_ + i) * (long)N_;
  float sc[H_];
#pragma unroll
  for (int h = 0; h < H_; ++h) sc[h] = sctx[((long)(b * H_ + h)) * N_ + i];
  float mx[H_];
#pragma unroll
  for (int h = 0; h < H_; ++h) mx[h] = -3.4e38f;
  for (int j = t; j < N_; j += 256) {
    int av = adjRow[j];
#pragma unroll
    for (int h = 0; h < H_; ++h) {
      float e = sc[h] + sh[((long)(b * H_ + h)) * N_ + j];
      e = e > 0.f ? e : ALPHA_ * e;
      float val = av > 0 ? e : NEG_BIG_;
      mx[h] = fmaxf(mx[h], val);
    }
  }
  for (int h = 0; h < H_; ++h) {
    float m = blockReduceMax256(mx[h], red);
    if (t == 0) mSh[h] = m;
  }
  __syncthreads();
  float sm[H_];
#pragma unroll
  for (int h = 0; h < H_; ++h) sm[h] = 0.f;
  for (int j = t; j < N_; j += 256) {
    int av = adjRow[j];
#pragma unroll
    for (int h = 0; h < H_; ++h) {
      float e = sc[h] + sh[((long)(b * H_ + h)) * N_ + j];
      e = e > 0.f ? e : ALPHA_ * e;
      float val = av > 0 ? e : NEG_BIG_;
      sm[h] += expf(val - mSh[h]);
    }
  }
  for (int h = 0; h < H_; ++h) {
    float l = blockReduceSum256(sm[h], red);
    if (t == 0) {
      long idx = ((long)(b * H_ + h)) * N_ + i;
      mg[idx] = mSh[h]; lg[idx] = l;
    }
  }
}

// hp = att @ Ws_ctx  (p built directly in WMMA A-register layout), leaky,
// scattered into gat[b][n][C].  One wave per (bh, 16-row tile).
__global__ void gat_acc_kernel(const int* adj, const float* sctx, const float* sh,
                               const float* mg, const float* lg,
                               const __bf16* wsctxT, float* gat) {
  int bh = blockIdx.y; int b = bh / H_, h = bh % H_;
  int i0 = blockIdx.x * 16;
  int lane = threadIdx.x & 31;
  int m = lane & 15, hi = (lane >> 4) & 1;
  int r = i0 + m;
  long bhN = (long)bh * N_;
  float sc = sctx[bhN + r];
  float mm = mg[bhN + r];
  float li = 1.0f / lg[bhN + r];
  const int* adjRow = adj + ((long)b * N_ + r) * (long)N_;
  const float* shp = sh + bhN;
  const __bf16* B0 = wsctxT + (long)bh * FH_ * N_;
  v8f c0 = {}, c1 = {};
  for (int j0 = 0; j0 < N_; j0 += 32) {
    v16bf a;
#pragma unroll
    for (int e = 0; e < 16; ++e) {
      int K = (e < 8) ? (e + 8 * hi) : (16 + (e - 8) + 8 * hi);  // A layout
      int j = j0 + K;
      float ev = sc + shp[j];
      ev = ev > 0.f ? ev : ALPHA_ * ev;
      float val = (adjRow[j] > 0) ? ev : NEG_BIG_;
      a[e] = f2bf(expf(val - mm) * li);                          // normalized p
    }
    v16bf b0 = ldB_bf16(B0,  0, N_, j0, lane);  // f = 0..15
    v16bf b1 = ldB_bf16(B0, 16, N_, j0, lane);  // f = 16..31
    c0 = wmma_bf16(a, b0, c0);
    c1 = wmma_bf16(a, b1, c1);
  }
  int fc = m;
#pragma unroll
  for (int v = 0; v < 8; ++v) {
    int row = i0 + v + hi * 8;
    float x0 = c0[v]; x0 = x0 > 0.f ? x0 : NSLOPE_ * x0;
    float x1 = c1[v]; x1 = x1 > 0.f ? x1 : NSLOPE_ * x1;
    long base = ((long)b * N_ + row) * C_ + h * FH_;
    gat[base + fc] = x0;
    gat[base + 16 + fc] = x1;
  }
}

// Row (axis-2) norm, ddof=1; optional elementwise add of a second input.
__global__ void rownorm_kernel(const float* in0, const float* in1, float* out) {
  long row = blockIdx.x; int c = threadIdx.x;
  __shared__ float red[256];
  __shared__ float mv, sv;
  float v = in0[row * C_ + c];
  if (in1) v += in1[row * C_ + c];
  float s = blockReduceSum256(v, red);
  if (c == 0) mv = s * (1.0f / C_);
  __syncthreads();
  float d = v - mv;
  float s2 = blockReduceSum256(d * d, red);
  if (c == 0) sv = sqrtf(s2 / (C_ - 1));
  __syncthreads();
  out[row * C_ + c] = d / (sv + EPS_);
}

// Column (axis-1) norm over N per (b, channel), ddof=1.  32ch x 8 row-groups.
__global__ void colnorm_kernel(const float* in, float* out) {
  int b = blockIdx.y;
  int ch = blockIdx.x * 32 + (threadIdx.x & 31);
  int rg = threadIdx.x >> 5;
  int li = threadIdx.x & 31;
  __shared__ float s1[8][32], s2[8][32];
  __shared__ float mS[32], sS[32];
  float a = 0.f, q = 0.f;
  for (int n = rg; n < N_; n += 8) {
    float v = in[((long)b * N_ + n) * C_ + ch];
    a += v; q += v * v;
  }
  s1[rg][li] = a; s2[rg][li] = q;
  __syncthreads();
  if (rg == 0) {
    float t1 = 0.f, t2 = 0.f;
#pragma unroll
    for (int r2 = 0; r2 < 8; ++r2) { t1 += s1[r2][li]; t2 += s2[r2][li]; }
    float mean = t1 * (1.0f / N_);
    float var = (t2 - (float)N_ * mean * mean) / (float)(N_ - 1);
    mS[li] = mean; sS[li] = sqrtf(fmaxf(var, 0.f));
  }
  __syncthreads();
  float mean = mS[li], sd = sS[li];
  for (int n = rg; n < N_; n += 8) {
    long idx = ((long)b * N_ + n) * C_ + ch;
    out[idx] = (in[idx] - mean) / (sd + EPS_);
  }
}

// out[n][co] = sum_k A[n][k]*W[co][k] + bias[co].  4096x256x256, bf16 WMMA.
// mode 0: fp32 row-major.  mode 1: bf16 row-major.  mode 2: bf16 vT[b][h][d][n].
__global__ void gemm_bias_kernel(const float* A, const float* W, const float* bias,
                                 int mode, float* outF, __bf16* outB) {
  int lane = threadIdx.x & 31;
  int tile = blockIdx.x * 4 + (threadIdx.x >> 5);
  int rt = tile >> 4, ct = tile & 15;           // 256 row-tiles x 16 col-tiles
  int r0 = rt * 16, c0 = ct * 16;
  if (r0 >= B_ * N_) return;
  v8f acc = {};
  for (int k0 = 0; k0 < C_; k0 += 32) {
    if (k0 + 32 < C_)
      __builtin_prefetch(A + (long)(r0 + (lane & 15)) * C_ + k0 + 32, 0, 1);
    v16bf a = ldA_f32(A, r0, C_, k0, lane);
    v16bf b = ldB_f32(W, c0, C_, k0, lane);     // W row co = B column co
    acc = wmma_bf16(a, b, acc);
  }
  int hi = (lane >> 4) & 1, m = lane & 15;
  int co = c0 + m;
  float bv = bias ? bias[co] : 0.f;
  if (mode == 0) {
#pragma unroll
    for (int v = 0; v < 8; ++v) {
      int row = r0 + v + hi * 8;
      outF[(long)row * C_ + co] = acc[v] + bv;
    }
  } else if (mode == 1) {
#pragma unroll
    for (int v = 0; v < 8; ++v) {
      int row = r0 + v + hi * 8;
      outB[(long)row * C_ + co] = f2bf(acc[v] + bv);
    }
  } else {                                      // head-transposed V
    int hh = co / DH_, d = co % DH_;
#pragma unroll
    for (int v = 0; v < 8; ++v) {
      int row = r0 + v + hi * 8;
      int bb = row >> 11, n = row & (N_ - 1);
      outB[(((long)bb * H_ + hh) * DH_ + d) * N_ + n] = f2bf(acc[v] + bv);
    }
  }
}

// MHA pass 1: per-row max & denom via score WMMAs (one 16x16x32 per tile).
__global__ void mha_ml_kernel(const __bf16* q, const __bf16* k, float* mM, float* lM) {
  int bh = blockIdx.y; int b = bh / H_, h = bh % H_;
  int i0 = blockIdx.x * 16;
  int lane = threadIdx.x & 31;
  int hi = (lane >> 4) & 1;
  const float scale = 0.17677669529663687f;     // 1/sqrt(32)
  const __bf16* qB = q + (long)b * N_ * C_ + h * DH_;
  const __bf16* kB = k + (long)b * N_ * C_ + h * DH_;
  v16bf aq = ldA_bf16(qB, i0, C_, 0, lane);
  float m8[8];
#pragma unroll
  for (int v = 0; v < 8; ++v) m8[v] = -3.4e38f;
  for (int j0 = 0; j0 < N_; j0 += 16) {
    v16bf bk = ldB_bf16(kB, j0, C_, 0, lane);
    v8f s = {};
    s = wmma_bf16(aq, bk, s);
#pragma unroll
    for (int v = 0; v < 8; ++v) m8[v] = fmaxf(m8[v], s[v] * scale);
  }
#pragma unroll
  for (int v = 0; v < 8; ++v) m8[v] = halfMax16(m8[v]);
  float l8[8];
#pragma unroll
  for (int v = 0; v < 8; ++v) l8[v] = 0.f;
  for (int j0 = 0; j0 < N_; j0 += 16) {
    v16bf bk = ldB_bf16(kB, j0, C_, 0, lane);
    v8f s = {};
    s = wmma_bf16(aq, bk, s);
#pragma unroll
    for (int v = 0; v < 8; ++v) l8[v] += expf(s[v] * scale - m8[v]);
  }
#pragma unroll
  for (int v = 0; v < 8; ++v) l8[v] = halfSum16(l8[v]);
  if ((lane & 15) == 0) {
#pragma unroll
    for (int v = 0; v < 8; ++v) {
      long idx = (long)bh * N_ + i0 + hi * 8 + v;
      mM[idx] = m8[v]; lM[idx] = l8[v];
    }
  }
}

// MHA pass 2: recompute scores, p = exp(s-m)/l, D-layout -> LDS -> A-layout,
// o += p @ v via WMMA.  4 waves/block share one (b,h): k/vT tiles staged in
// LDS by the Tensor Data Mover (double-buffered, TENSORcnt-synchronized).
__global__ void mha_acc_kernel(const __bf16* q, const __bf16* k, const __bf16* vT,
                               const float* mM, const float* lM, float* o) {
  __shared__ __bf16 sK[2][32][32];      // k rows j-local, cols d
  __shared__ __bf16 sV[2][32][32];      // vT rows d, cols j-local
  __shared__ __bf16 pS[4][16][32];      // per-wave p staging
  int w = threadIdx.x >> 5;
  int lane = threadIdx.x & 31;
  int bh = blockIdx.y; int b = bh / H_, h = bh % H_;
  int i0 = (blockIdx.x * 4 + w) * 16;
  int hi = (lane >> 4) & 1, m = lane & 15;
  const float scale = 0.17677669529663687f;
  const __bf16* qB = q + (long)b * N_ * C_ + h * DH_;
  const __bf16* kB = k + (long)b * N_ * C_ + h * DH_;
  const __bf16* vB = vT + (long)bh * DH_ * N_;
  v16bf aq = ldA_bf16(qB, i0, C_, 0, lane);
  float mh[8], li[8];
#pragma unroll
  for (int v = 0; v < 8; ++v) {
    long idx = (long)bh * N_ + i0 + hi * 8 + v;
    mh[v] = mM[idx]; li[v] = 1.0f / lM[idx];
  }
  v8f o0 = {}, o1 = {};
  constexpr int T = N_ / 32;
#if HAVE_TDM
  if (w == 0) {   // wave 0 drives the TDM (TENSORcnt is per-wave, in-order)
    tdm_load_2d_bf16(lds_addr_of(&sK[0][0][0]), kB, 32, 32, 32, N_, C_);
    tdm_load_2d_bf16(lds_addr_of(&sV[0][0][0]), vB, 32, 32, N_, 32, N_);
  }
#endif
  for (int t = 0; t < T; ++t) {
    int cur = t & 1;
    int j0 = t * 32;
#if HAVE_TDM
    if (w == 0) {
      if (t + 1 < T) {
        int jn = j0 + 32;
        tdm_load_2d_bf16(lds_addr_of(&sK[cur ^ 1][0][0]), kB + (long)jn * C_,
                         32, 32, 32, (unsigned)(N_ - jn), C_);
        tdm_load_2d_bf16(lds_addr_of(&sV[cur ^ 1][0][0]), vB + jn,
                         32, 32, (unsigned)(N_ - jn), 32, N_);
        __builtin_amdgcn_s_wait_tensorcnt(2);   // chunk t landed (in-order)
      } else {
        __builtin_amdgcn_s_wait_tensorcnt(0);
      }
    }
#else
    coop_copy_tile32(&sK[cur][0][0], kB + (long)j0 * C_, C_);
    coop_copy_tile32(&sV[cur][0][0], vB + j0, N_);
#endif
    __syncthreads();                            // tiles visible to all waves
    const __bf16* sk = &sK[cur][0][0];
    const __bf16* sv = &sV[cur][0][0];
    v16bf bk0 = ldB_bf16(sk, 0, 32, 0, lane);   // j-local 0..15
    v16bf bk1 = ldB_bf16(sk, 16, 32, 0, lane);  // j-local 16..31
    v8f s0 = {}, s1 = {};
    s0 = wmma_bf16(aq, bk0, s0);
    s1 = wmma_bf16(aq, bk1, s1);
#pragma unroll
    for (int v = 0; v < 8; ++v) {
      int rr = v + hi * 8;
      pS[w][rr][m]      = f2bf(expf(s0[v] * scale - mh[v]) * li[v]);
      pS[w][rr][16 + m] = f2bf(expf(s1[v] * scale - mh[v]) * li[v]);
    }
    // per-wave LDS slice; DS ops from one wave stay in-order
    v16bf ap;
#pragma unroll
    for (int e = 0; e < 16; ++e) {
      int K = (e < 8) ? (e + 8 * hi) : (16 + (e - 8) + 8 * hi);  // A layout
      ap[e] = pS[w][m][K];
    }
    v16bf bv0 = ldB_bf16(sv, 0, 32, 0, lane);   // d = 0..15
    v16bf bv1 = ldB_bf16(sv, 16, 32, 0, lane);  // d = 16..31
    o0 = wmma_bf16(ap, bv0, o0);
    o1 = wmma_bf16(ap, bv1, o1);
    __syncthreads();                            // done with buf before reissue
  }
#pragma unroll
  for (int v = 0; v < 8; ++v) {
    int row = i0 + v + hi * 8;
    long base = ((long)b * N_ + row) * C_ + h * DH_;
    o[base + m]      = o0[v];
    o[base + 16 + m] = o1[v];
  }
}

// ----------------------------------------------------------------- launch ---

extern "C" void kernel_launch(void* const* d_in, const int* in_sizes, int n_in,
                              void* d_out, int out_size, void* d_ws, size_t ws_size,
                              hipStream_t stream) {
  (void)in_sizes; (void)n_in; (void)out_size; (void)ws_size;
  const float* x    = (const float*)d_in[0];
  const float* ctxp = (const float*)d_in[1];
  const float* ctx  = (const float*)d_in[2];
  /* d_in[3] = src : unused by the reference */
  const int*   adj  = (const int*)d_in[4];
  const float* Ws   = (const float*)d_in[5];
  const float* Wt   = (const float*)d_in[6];
  const float* av   = (const float*)d_in[7];
  const float* Wlin = (const float*)d_in[8];
  const float* blin = (const float*)d_in[9];
  const float* Win  = (const float*)d_in[10];
  const float* bin  = (const float*)d_in[11];
  const float* Wout = (const float*)d_in[12];
  const float* bout = (const float*)d_in[13];
  float* out = (float*)d_out;

  // workspace carve-up (~37 MB total)
  char* wbase = (char*)d_ws; size_t off = 0;
  auto alloc = [&](size_t bytes) -> void* {
    void* p = wbase + off; off += (bytes + 255) & ~(size_t)255; return p;
  };
  float*  wsa     = (float*)alloc((size_t)H_ * C_ * 4);
  float*  wta     = (float*)alloc((size_t)H_ * C_ * 4);
  float*  sctx    = (float*)alloc((size_t)B_ * H_ * N_ * 4);
  float*  sh      = (float*)alloc((size_t)B_ * H_ * N_ * 4);
  float*  mg      = (float*)alloc((size_t)B_ * H_ * N_ * 4);
  float*  lg      = (float*)alloc((size_t)B_ * H_ * N_ * 4);
  __bf16* wsctxT  = (__bf16*)alloc((size_t)B_ * H_ * FH_ * N_ * 2);
  float*  gat     = (float*)alloc((size_t)B_ * N_ * C_ * 4);
  float*  hA      = (float*)alloc((size_t)B_ * N_ * C_ * 4);
  float*  hB      = (float*)alloc((size_t)B_ * N_ * C_ * 4);
  float*  h1      = (float*)alloc((size_t)B_ * N_ * C_ * 4);
  float*  h1r     = (float*)alloc((size_t)B_ * N_ * C_ * 4);
  float*  h1n     = (float*)alloc((size_t)B_ * N_ * C_ * 4);
  __bf16* qb      = (__bf16*)alloc((size_t)B_ * N_ * C_ * 2);
  __bf16* kb      = (__bf16*)alloc((size_t)B_ * N_ * C_ * 2);
  __bf16* vTb     = (__bf16*)alloc((size_t)B_ * N_ * C_ * 2);
  float*  mM      = (float*)alloc((size_t)B_ * H_ * N_ * 4);
  float*  lM      = (float*)alloc((size_t)B_ * H_ * N_ * 4);
  float*  obuf    = (float*)alloc((size_t)B_ * N_ * C_ * 4);

  // 1) fold attention vector `a` into per-head channel vectors
  fold_a_kernel<<<(H_ * C_ + 255) / 256, 256, 0, stream>>>(Ws, Wt, av, wsa, wta);
  // 2) rank-1 logit components s_ctx, s_h
  score_vec_kernel<<<(B_ * H_ * N_ + 255) / 256, 256, 0, stream>>>(ctxp, x, wsa, wta, sctx, sh);
  // 3) Ws_ctx projection (WMMA), stored f-major bf16
  gat_proj_kernel<<<dim3((N_ / 16) * (FH_ / 16), B_ * H_), 32, 0, stream>>>(ctxp, Ws, wsctxT);
  // 4) GAT softmax statistics
  gat_ml_kernel<<<dim3(N_, B_), 256, 0, stream>>>(adj, sctx, sh, mg, lg);
  // 5) GAT aggregation hp = att @ Ws_ctx (WMMA) + leaky + head-interleave
  gat_acc_kernel<<<dim3(N_ / 16, B_ * H_), 32, 0, stream>>>(adj, sctx, sh, mg, lg, wsctxT, gat);
  // 6) h = double_norm(x + gat)
  rownorm_kernel<<<B_ * N_, 256, 0, stream>>>(x, gat, hA);
  colnorm_kernel<<<dim3(C_ / 32, B_), 256, 0, stream>>>(hA, hB);
  // 7) h1 = h @ W_lin.T + b_lin   (WMMA GEMM)
  gemm_bias_kernel<<<(B_ * N_ / 16) * (C_ / 16) / 4, 128, 0, stream>>>(hB, Wlin, blin, 0, h1, nullptr);
  // 8) h = double_norm(h1)
  rownorm_kernel<<<B_ * N_, 256, 0, stream>>>(h1, nullptr, h1r);
  colnorm_kernel<<<dim3(C_ / 32, B_), 256, 0, stream>>>(h1r, h1n);
  // 9) QKV projections (WMMA GEMMs); v written head-transposed
  gemm_bias_kernel<<<(B_ * N_ / 16) * (C_ / 16) / 4, 128, 0, stream>>>(h1n, Win,              bin,          1, nullptr, qb);
  gemm_bias_kernel<<<(B_ * N_ / 16) * (C_ / 16) / 4, 128, 0, stream>>>(ctxp, Win + C_ * C_,     bin + C_,     1, nullptr, kb);
  gemm_bias_kernel<<<(B_ * N_ / 16) * (C_ / 16) / 4, 128, 0, stream>>>(ctx,  Win + 2 * C_ * C_, bin + 2 * C_, 2, nullptr, vTb);
  // 10) MHA softmax statistics then normalized accumulation (WMMA both)
  mha_ml_kernel<<<dim3(N_ / 16, B_ * H_), 32, 0, stream>>>(qb, kb, mM, lM);
  mha_acc_kernel<<<dim3(N_ / 16 / 4, B_ * H_), 128, 0, stream>>>(qb, kb, vTb, mM, lM, obuf);
  // 11) output projection (WMMA GEMM) -> d_out
  gemm_bias_kernel<<<(B_ * N_ / 16) * (C_ / 16) / 4, 128, 0, stream>>>(obuf, Wout, bout, 0, out, nullptr);
}
